// MLA_76682346102857
// MI455X (gfx1250) — compile-verified, matrix operand order checked by reading, not run on hardware
//
#include <hip/hip_runtime.h>
#include <hip/hip_bf16.h>

// ---------------------------------------------------------------------------
// MLA forward for MI455X (gfx1250, wave32, WMMA, async global->LDS).
// All intermediates are f16 (fp32 accumulate in WMMA), halving HBM/L2 bytes:
//   0) convert x + weights to f16
//   1) qa   = xh @ wq_a + b          (4096 x 512,  K=2048)  f16 out
//   2) q    = qa @ wq_b + b          (4096 x 3072, K=512)   f16 out
//   3) ckv  = xh @ wkv_a + b         (4096 x 576,  K=2048)  f16 out
//   4) kv   = ckv[:, :512] @ wkv_b   (4096 x 4096, K=512, lda=576)
//   5) pack: Q (scaled+roped), K (roped pe broadcast), V^T   (f16)
//   6) flash attention (online softmax, WMMA f16, fp32 accum) -> f16
//   7) out  = attn @ wo + b          (4096 x 2048, K=2048)  fp32 out
// GEMM: 128x64x32 block tile, 8 waves x (32x32), double-buffered LDS with
// GLOBAL_LOAD_ASYNC_TO_LDS_B128 staging of the A tile (ASYNCcnt pipeline).
// ---------------------------------------------------------------------------

typedef __attribute__((ext_vector_type(16))) _Float16 v16h;
typedef __attribute__((ext_vector_type(8)))  _Float16 v8h;
typedef __attribute__((ext_vector_type(8)))  float    v8f;
typedef __attribute__((ext_vector_type(4)))  int      v4i;

union Frag {
    v16h v;
    v8h  h[2];
};

static __device__ __forceinline__ v8f wmma_f16(const v16h& a, const v16h& b, const v8f& c) {
    return __builtin_amdgcn_wmma_f32_16x16x32_f16(false, a, false, b, (short)0, c, false, false);
}

// ---- async global->LDS copy (16 bytes / lane), ASYNCcnt tracked ----------
#if __has_builtin(__builtin_amdgcn_global_load_async_to_lds_b128)
#define MLA_HAS_ASYNC_LDS 1
#else
#define MLA_HAS_ASYNC_LDS 0
#endif

static __device__ __forceinline__ void copy16_g2l(_Float16* lds, const _Float16* g) {
#if MLA_HAS_ASYNC_LDS
    // addrspacecast first (generic -> specific), then pointee reinterpret
    const __attribute__((address_space(1))) _Float16* g1 =
        (const __attribute__((address_space(1))) _Float16*)g;
    __attribute__((address_space(3))) _Float16* l3 =
        (__attribute__((address_space(3))) _Float16*)lds;
    __builtin_amdgcn_global_load_async_to_lds_b128(
        (__attribute__((address_space(1))) v4i*)g1,
        (__attribute__((address_space(3))) v4i*)l3, 0, 0);
#else
    *(v8h*)lds = *(const v8h*)g;
#endif
}

static __device__ __forceinline__ void async_wait_le2() {
#if MLA_HAS_ASYNC_LDS
#if __has_builtin(__builtin_amdgcn_s_wait_asynccnt)
    __builtin_amdgcn_s_wait_asynccnt(2);
#else
    asm volatile("s_wait_asynccnt 0x2" ::: "memory");
#endif
#endif
}

static __device__ __forceinline__ void async_wait_0() {
#if MLA_HAS_ASYNC_LDS
#if __has_builtin(__builtin_amdgcn_s_wait_asynccnt)
    __builtin_amdgcn_s_wait_asynccnt(0);
#else
    asm volatile("s_wait_asynccnt 0x0" ::: "memory");
#endif
#endif
}

// ---------------------------------------------------------------------------
// fp32 -> f16 conversion (8 elements / thread; all sizes are multiples of 8)
// ---------------------------------------------------------------------------
__global__ __launch_bounds__(256) void mla_cvt_f16(
    const float* __restrict__ s, _Float16* __restrict__ d, int n8)
{
    const int i = blockIdx.x * 256 + threadIdx.x;
    if (i < n8) {
        const float4 f0 = *(const float4*)(s + (size_t)i * 8);
        const float4 f1 = *(const float4*)(s + (size_t)i * 8 + 4);
        v8h h;
        h[0] = (_Float16)f0.x; h[1] = (_Float16)f0.y;
        h[2] = (_Float16)f0.z; h[3] = (_Float16)f0.w;
        h[4] = (_Float16)f1.x; h[5] = (_Float16)f1.y;
        h[6] = (_Float16)f1.z; h[7] = (_Float16)f1.w;
        *(v8h*)(d + (size_t)i * 8) = h;
    }
}

// ---------------------------------------------------------------------------
// Tiled GEMM: C[M,N] = A[M,K] @ B[K,N] + bias[N]
// f16 in, CT out (f16 or fp32), fp32 WMMA accumulate.
// Block: 256 threads (8 waves), tile 128x64, K-step 32, double-buffered LDS.
// A tile staged with async global->LDS b128; B staged transposed [n][k].
// Wave w: rows (w>>1)*32, cols (w&1)*32; 4 accumulators, 4 WMMAs / K-step.
// ---------------------------------------------------------------------------
template <typename CT>
__global__ __launch_bounds__(256) void mla_gemm_h(
    const _Float16* __restrict__ A, const _Float16* __restrict__ B,
    const float* __restrict__ bias, CT* __restrict__ C,
    int M, int N, int K, int lda, int ldb, int ldc)
{
    __shared__ __align__(16) _Float16 As[2][128][40];  // [m][k], +8 pad (20 KB)
    __shared__ __align__(16) _Float16 Bs[2][64][40];   // [n][k] transposed (10 KB)

    (void)M;
    const int tid  = threadIdx.x;
    const int m0   = blockIdx.y * 128;
    const int n0   = blockIdx.x * 64;
    const int wave = tid >> 5;
    const int lane = tid & 31;
    const int lrow = lane & 15;
    const int lhi  = lane >> 4;
    const int kb   = lhi * 8;
    const int wm   = (wave >> 1) * 32;
    const int wn   = (wave & 1) * 32;

    // staging assignments
    const int bk   = tid >> 3;          // 0..31 (B k-row)
    const int bseg = (tid & 7) * 8;     // 0..56 (B n-seg)

    v8f acc00 = {}, acc01 = {}, acc10 = {}, acc11 = {};

    auto stageA = [&](int k0, int buf) {
        #pragma unroll
        for (int i = 0; i < 2; ++i) {
            const int ch  = tid + i * 256;      // 0..511
            const int row = ch >> 2;
            const int seg = (ch & 3) * 8;
            copy16_g2l(&As[buf][row][seg],
                       A + (size_t)(m0 + row) * lda + k0 + seg);
        }
    };
    auto stageB = [&](int k0, int buf) {
        const _Float16* bp = B + (size_t)(k0 + bk) * ldb + n0 + bseg;
        const v8h bv = *(const v8h*)bp;
        if (k0 + 32 < K) __builtin_prefetch(bp + (size_t)32 * ldb, 0, 1);
        #pragma unroll
        for (int j = 0; j < 8; ++j) Bs[buf][bseg + j][bk] = bv[j];
    };

    stageA(0, 0);
    stageB(0, 0);

    const int T = K >> 5;
    for (int t = 0; t < T; ++t) {
        const int cur = t & 1;
        if (t + 1 < T) {
            stageA((t + 1) * 32, 1 - cur);
            stageB((t + 1) * 32, 1 - cur);
            async_wait_le2();               // tile t's 2 async ops retired
        } else {
            async_wait_0();
        }
        __syncthreads();

        Frag a0, a1, b0, b1;
        a0.h[0] = *(const v8h*)&As[cur][wm + lrow][kb];
        a0.h[1] = *(const v8h*)&As[cur][wm + lrow][kb + 16];
        a1.h[0] = *(const v8h*)&As[cur][wm + 16 + lrow][kb];
        a1.h[1] = *(const v8h*)&As[cur][wm + 16 + lrow][kb + 16];
        b0.h[0] = *(const v8h*)&Bs[cur][wn + lrow][kb];
        b0.h[1] = *(const v8h*)&Bs[cur][wn + lrow][kb + 16];
        b1.h[0] = *(const v8h*)&Bs[cur][wn + 16 + lrow][kb];
        b1.h[1] = *(const v8h*)&Bs[cur][wn + 16 + lrow][kb + 16];

        acc00 = wmma_f16(a0.v, b0.v, acc00);
        acc01 = wmma_f16(a0.v, b1.v, acc01);
        acc10 = wmma_f16(a1.v, b0.v, acc10);
        acc11 = wmma_f16(a1.v, b1.v, acc11);

        __syncthreads();
    }

    // epilogue: (vgpr r, lane l): m = wm + r + 8*(l>>4) (+16), n = wn + (l&15) (+16)
    const int ncol = n0 + wn + lrow;
    const float bv0 = bias[ncol];
    const float bv1 = bias[ncol + 16];
    #pragma unroll
    for (int r = 0; r < 8; ++r) {
        const int row0 = m0 + wm + r + 8 * lhi;
        const int row1 = row0 + 16;
        C[(size_t)row0 * ldc + ncol]      = (CT)(acc00[r] + bv0);
        C[(size_t)row0 * ldc + ncol + 16] = (CT)(acc01[r] + bv1);
        C[(size_t)row1 * ldc + ncol]      = (CT)(acc10[r] + bv0);
        C[(size_t)row1 * ldc + ncol + 16] = (CT)(acc11[r] + bv1);
    }
}

// ---------------------------------------------------------------------------
// Pack kernel: build Q/K (roped) and V^T from f16 GEMM outputs (fp32 math).
//   Qh[b*16+h][s][0..192) : nope | rope(q_pe), all * 1/sqrt(128)
//   Kh[b*16+h][s][0..192) : kv_nope | rope(k_pe) broadcast over h
//   Vt[b*16+h][d][s]      : V transposed for contiguous-key WMMA B fragments
// ---------------------------------------------------------------------------
__global__ __launch_bounds__(256) void mla_pack_rope(
    const _Float16* __restrict__ qfull,   // [4096][3072]
    const _Float16* __restrict__ kvfull,  // [4096][4096]
    const _Float16* __restrict__ ckv,     // [4096][576]
    _Float16* __restrict__ Qh,            // [32][2048][192]
    _Float16* __restrict__ Kh,            // [32][2048][192]
    _Float16* __restrict__ Vt)            // [32][128][2048]
{
    const int row = blockIdx.x;       // 0..4095
    const int b   = row >> 11;
    const int s   = row & 2047;
    const int tid = threadIdx.x;
    const float scale = 0.08838834764831845f;   // 1/sqrt(128)

    // nope part of Q, K and V
    for (int idx = tid; idx < 16 * 128; idx += 256) {
        const int h = idx >> 7;
        const int d = idx & 127;
        const float qv = (float)qfull[(size_t)row * 3072 + h * 192 + d];
        const _Float16 kn = kvfull[(size_t)row * 4096 + h * 256 + d];
        const _Float16 vv = kvfull[(size_t)row * 4096 + h * 256 + 128 + d];
        const size_t qkb = (((size_t)(b * 16 + h)) * 2048 + s) * 192 + d;
        Qh[qkb] = (_Float16)(qv * scale);
        Kh[qkb] = kn;
        Vt[(((size_t)(b * 16 + h)) * 128 + d) * 2048 + s] = vv;
    }

    // rope part of Q: 16 heads x 32 complex pairs
    for (int idx = tid; idx < 16 * 32; idx += 256) {
        const int h = idx >> 5;
        const int i = idx & 31;
        const float inv = __powf(1000000.0f, -(float)i / 32.0f);
        const float ang = (float)s * inv;
        const float c = __cosf(ang), sn = __sinf(ang);
        const float xr = (float)qfull[(size_t)row * 3072 + h * 192 + 128 + 2 * i];
        const float xi = (float)qfull[(size_t)row * 3072 + h * 192 + 128 + 2 * i + 1];
        const size_t base = (((size_t)(b * 16 + h)) * 2048 + s) * 192 + 128 + 2 * i;
        Qh[base]     = (_Float16)((xr * c - xi * sn) * scale);
        Qh[base + 1] = (_Float16)((xr * sn + xi * c) * scale);
    }

    // rope part of K (shared across heads): 32 pairs, broadcast to 16 heads
    for (int i = tid; i < 32; i += 256) {
        const float inv = __powf(1000000.0f, -(float)i / 32.0f);
        const float ang = (float)s * inv;
        const float c = __cosf(ang), sn = __sinf(ang);
        const float xr = (float)ckv[(size_t)row * 576 + 512 + 2 * i];
        const float xi = (float)ckv[(size_t)row * 576 + 512 + 2 * i + 1];
        const _Float16 o0 = (_Float16)(xr * c - xi * sn);
        const _Float16 o1 = (_Float16)(xr * sn + xi * c);
        #pragma unroll
        for (int h = 0; h < 16; ++h) {
            const size_t base = (((size_t)(b * 16 + h)) * 2048 + s) * 192 + 128 + 2 * i;
            Kh[base]     = o0;
            Kh[base + 1] = o1;
        }
    }
}

// ---------------------------------------------------------------------------
// Flash attention: block = (b*16+h, q-tile of 64). 128 threads = 4 waves,
// wave w owns q-rows [w*16, w*16+16). Keys processed in chunks of 32 with
// online softmax; S and O updates via v_wmma_f32_16x16x32_f16 (20 / chunk).
// ---------------------------------------------------------------------------
__global__ __launch_bounds__(128) void mla_flash_attn(
    const _Float16* __restrict__ Qh,   // [32][2048][192] (pre-scaled)
    const _Float16* __restrict__ Kh,   // [32][2048][192]
    const _Float16* __restrict__ Vt,   // [32][128][2048]
    _Float16* __restrict__ attn)       // [4096][2048] = [b*2048+s][h*128+d]
{
    __shared__ __align__(16) _Float16 Qs[64][200];      // q tile, +8 pad
    __shared__ __align__(16) _Float16 Ps[4][16][40];    // per-wave P staging

    const int bh   = blockIdx.x;        // b*16+h
    const int q0   = blockIdx.y * 64;
    const int tid  = threadIdx.x;
    const int wave = tid >> 5;
    const int lane = tid & 31;
    const int lrow = lane & 15;
    const int lhi  = lane >> 4;
    const int kb   = lhi * 8;
    const int wm   = wave * 16;

    // stage Q tile (64 x 192) into LDS
    const _Float16* Qbase = Qh + ((size_t)bh * 2048 + q0) * 192;
    for (int idx = tid; idx < 64 * 24; idx += 128) {
        const int r = idx / 24;
        const int cseg = (idx % 24) * 8;
        *(v8h*)&Qs[r][cseg] = *(const v8h*)(Qbase + (size_t)r * 192 + cseg);
    }
    __syncthreads();

    // preload the 6 Q fragments (d-chunks of 32) for this wave's 16 rows
    Frag qf[6];
    #pragma unroll
    for (int c = 0; c < 6; ++c) {
        qf[c].h[0] = *(const v8h*)&Qs[wm + lrow][c * 32 + kb];
        qf[c].h[1] = *(const v8h*)&Qs[wm + lrow][c * 32 + kb + 16];
    }

    float mrow[8], lsum[8];
    v8f o[8];
    #pragma unroll
    for (int r = 0; r < 8; ++r) { mrow[r] = -1e30f; lsum[r] = 0.0f; }
    #pragma unroll
    for (int d = 0; d < 8; ++d) { v8f z = {}; o[d] = z; }

    const _Float16* Kb = Kh + (size_t)bh * 2048 * 192;
    const _Float16* Vb = Vt + (size_t)bh * 128 * 2048;

    for (int key0 = 0; key0 < 2048; key0 += 32) {
        // ---- S = Q @ K^T for this 32-key chunk (two 16-key column blocks) ----
        v8f s0 = {};
        v8f s1 = {};
        #pragma unroll
        for (int c = 0; c < 6; ++c) {
            Frag bf0, bf1;
            const _Float16* kp0 = Kb + (size_t)(key0 + lrow) * 192 + c * 32 + kb;
            const _Float16* kp1 = Kb + (size_t)(key0 + 16 + lrow) * 192 + c * 32 + kb;
            bf0.h[0] = *(const v8h*)kp0;
            bf0.h[1] = *(const v8h*)(kp0 + 16);
            bf1.h[0] = *(const v8h*)kp1;
            bf1.h[1] = *(const v8h*)(kp1 + 16);
            s0 = wmma_f16(qf[c].v, bf0.v, s0);
            s1 = wmma_f16(qf[c].v, bf1.v, s1);
        }

        // ---- online softmax (row m = wm + r + 8*lhi) ----
        #pragma unroll
        for (int r = 0; r < 8; ++r) {
            float v = fmaxf(s0[r], s1[r]);
            v = fmaxf(v, __shfl_xor(v, 8, 32));
            v = fmaxf(v, __shfl_xor(v, 4, 32));
            v = fmaxf(v, __shfl_xor(v, 2, 32));
            v = fmaxf(v, __shfl_xor(v, 1, 32));
            const float mnew  = fmaxf(mrow[r], v);
            const float alpha = __expf(mrow[r] - mnew);
            mrow[r] = mnew;

            const float p0 = __expf(s0[r] - mnew);
            const float p1 = __expf(s1[r] - mnew);
            float psum = p0 + p1;
            psum += __shfl_xor(psum, 8, 32);
            psum += __shfl_xor(psum, 4, 32);
            psum += __shfl_xor(psum, 2, 32);
            psum += __shfl_xor(psum, 1, 32);
            lsum[r] = lsum[r] * alpha + psum;

            const int prow = r + 8 * lhi;
            Ps[wave][prow][lrow]      = (_Float16)p0;
            Ps[wave][prow][lrow + 16] = (_Float16)p1;

            #pragma unroll
            for (int d = 0; d < 8; ++d) o[d][r] *= alpha;
        }
        __syncthreads();   // make every lane's P slice visible (uniform loop)

        // P fragment: A-matrix 16(m) x 32(key)
        Frag pf;
        pf.h[0] = *(const v8h*)&Ps[wave][lrow][kb];
        pf.h[1] = *(const v8h*)&Ps[wave][lrow][kb + 16];

        // ---- O += P @ V  (8 d-blocks of 16) ----
        #pragma unroll
        for (int db = 0; db < 8; ++db) {
            Frag vf;
            const _Float16* vp = Vb + (size_t)(db * 16 + lrow) * 2048 + key0 + kb;
            vf.h[0] = *(const v8h*)vp;
            vf.h[1] = *(const v8h*)(vp + 16);
            o[db] = wmma_f16(pf.v, vf.v, o[db]);
        }
    }

    // ---- normalize, store f16: attn[(b*2048+q)][h*128+d] ----
    const int b = bh >> 4;
    const int h = bh & 15;
    #pragma unroll
    for (int r = 0; r < 8; ++r) {
        const int qrow = q0 + wm + r + 8 * lhi;
        const float inv = 1.0f / lsum[r];
        const size_t base = ((size_t)(b * 2048 + qrow)) * 2048 + h * 128;
        #pragma unroll
        for (int db = 0; db < 8; ++db) {
            attn[base + db * 16 + lrow] = (_Float16)(o[db][r] * inv);
        }
    }
}

// ---------------------------------------------------------------------------
// Host-side launch
// ---------------------------------------------------------------------------
extern "C" void kernel_launch(void* const* d_in, const int* in_sizes, int n_in,
                              void* d_out, int out_size, void* d_ws, size_t ws_size,
                              hipStream_t stream) {
    (void)in_sizes; (void)n_in; (void)out_size; (void)ws_size;

    const float* x       = (const float*)d_in[0];   // [2][2048][2048]
    const float* wq_a_w  = (const float*)d_in[1];   // [2048][512]
    const float* wq_a_b  = (const float*)d_in[2];   // [512]
    const float* wq_b_w  = (const float*)d_in[3];   // [512][3072]
    const float* wq_b_b  = (const float*)d_in[4];   // [3072]
    const float* wkv_a_w = (const float*)d_in[5];   // [2048][576]
    const float* wkv_a_b = (const float*)d_in[6];   // [576]
    const float* wkv_b_w = (const float*)d_in[7];   // [512][4096]
    const float* wkv_b_b = (const float*)d_in[8];   // [4096]
    const float* wo_w    = (const float*)d_in[9];   // [2048][2048]
    const float* wo_b    = (const float*)d_in[10];  // [2048]
    float* out = (float*)d_out;                     // [2][2048][2048]

    const size_t ROWS = 4096;   // bs * seq

    // workspace layout (all f16)
    _Float16* xh    = (_Float16*)d_ws;                  // [4096][2048]
    _Float16* w1h   = xh   + ROWS * 2048;               // [2048][512]
    _Float16* w2h   = w1h  + (size_t)2048 * 512;        // [512][3072]
    _Float16* w3h   = w2h  + (size_t)512 * 3072;        // [2048][576]
    _Float16* w4h   = w3h  + (size_t)2048 * 576;        // [512][4096]
    _Float16* w5h   = w4h  + (size_t)512 * 4096;        // [2048][2048]
    _Float16* qa    = w5h  + (size_t)2048 * 2048;       // [4096][512]
    _Float16* qfull = qa   + ROWS * 512;                // [4096][3072]
    _Float16* ckv   = qfull + ROWS * 3072;              // [4096][576]
    _Float16* kv    = ckv  + ROWS * 576;                // [4096][4096]
    _Float16* attnb = kv   + ROWS * 4096;               // [4096][2048]
    _Float16* Qh    = attnb + ROWS * 2048;              // [32][2048][192]
    _Float16* Kh    = Qh   + (size_t)32 * 2048 * 192;   // [32][2048][192]
    _Float16* Vt    = Kh   + (size_t)32 * 2048 * 192;   // [32][128][2048]

    // 0) fp32 -> f16 conversions
    auto cvt = [&](const float* s, _Float16* d, size_t n) {
        const int n8 = (int)(n / 8);
        mla_cvt_f16<<<(n8 + 255) / 256, 256, 0, stream>>>(s, d, n8);
    };
    cvt(x,       xh,  ROWS * 2048);
    cvt(wq_a_w,  w1h, (size_t)2048 * 512);
    cvt(wq_b_w,  w2h, (size_t)512 * 3072);
    cvt(wkv_a_w, w3h, (size_t)2048 * 576);
    cvt(wkv_b_w, w4h, (size_t)512 * 4096);
    cvt(wo_w,    w5h, (size_t)2048 * 2048);

    // 1) qa = xh @ wq_a + b
    mla_gemm_h<_Float16><<<dim3(512 / 64, ROWS / 128), 256, 0, stream>>>(
        xh, w1h, wq_a_b, qa, (int)ROWS, 512, 2048, 2048, 512, 512);
    // 2) q = qa @ wq_b + b
    mla_gemm_h<_Float16><<<dim3(3072 / 64, ROWS / 128), 256, 0, stream>>>(
        qa, w2h, wq_b_b, qfull, (int)ROWS, 3072, 512, 512, 3072, 3072);
    // 3) ckv = xh @ wkv_a + b
    mla_gemm_h<_Float16><<<dim3(576 / 64, ROWS / 128), 256, 0, stream>>>(
        xh, w3h, wkv_a_b, ckv, (int)ROWS, 576, 2048, 2048, 576, 576);
    // 4) kv = ckv[:, :512] @ wkv_b + b   (lda = 576)
    mla_gemm_h<_Float16><<<dim3(4096 / 64, ROWS / 128), 256, 0, stream>>>(
        ckv, w4h, wkv_b_b, kv, (int)ROWS, 4096, 512, 576, 4096, 4096);
    // 5) pack + rope
    mla_pack_rope<<<dim3((int)ROWS), 256, 0, stream>>>(qfull, kv, ckv, Qh, Kh, Vt);
    // 6) flash attention
    mla_flash_attn<<<dim3(32, 2048 / 64), 128, 0, stream>>>(Qh, Kh, Vt, attnb);
    // 7) out = attn @ wo + b (fp32 out)
    mla_gemm_h<float><<<dim3(2048 / 64, ROWS / 128), 256, 0, stream>>>(
        attnb, w5h, wo_b, out, (int)ROWS, 2048, 2048, 2048, 2048, 2048);
}
